// DynamicOHEM_50173807952060
// MI455X (gfx1250) — compile-verified, matrix operand order checked by reading, not run on hardware
//
#include <hip/hip_runtime.h>
#include <hip/hip_bf16.h>

typedef __attribute__((ext_vector_type(16))) __bf16          v16bf;
typedef __attribute__((ext_vector_type(8)))  float           v8f;
typedef __attribute__((ext_vector_type(8)))  unsigned short  us8v;

union ABfrag { v16bf bf; us8v h[2]; };

// ---------- helpers ----------
__device__ __forceinline__ unsigned short f32_to_bf16(float f) {
    unsigned u = __float_as_uint(f);
    u += 0x7fffu + ((u >> 16) & 1u);   // round-to-nearest-even
    return (unsigned short)(u >> 16);
}
__device__ __forceinline__ unsigned keyOf(float f) {
    unsigned u = __float_as_uint(f);
    return (u & 0x80000000u) ? ~u : (u | 0x80000000u);   // monotone: bigger float -> bigger key
}
__device__ __forceinline__ float floatOf(unsigned k) {
    unsigned u = (k & 0x80000000u) ? (k & 0x7fffffffu) : ~k;
    return __uint_as_float(u);
}
__device__ __forceinline__ float rsum16(float v) {
    for (int o = 8; o >= 1; o >>= 1) v += __shfl_xor(v, o, 16);
    return v;
}

// ---------- kernel 0: pack W (fp32 -> bf16 B-fragments) + padded bias ----------
// B-fragment layout (symmetric to ISA A 16x32 layout): frag = ntile*KS + ks,
// 512 halves per frag; lane L holds column c = ntile*16 + (L&15); half h -> K offset
// kk = (h<8 ? h : h+8) + (L>=16 ? 8 : 0); k = ks*32 + kk.
// NT2 tiles (multiple of 4); padded tiles get W=0 and bias=-1e30 -> vanish in softmax.
__global__ void packW_kernel(const float* __restrict__ W, const float* __restrict__ bias,
                             unsigned short* __restrict__ wpk, float* __restrict__ biasPad,
                             int C, int NT2, int KS) {
    int e = blockIdx.x * 256 + threadIdx.x;
    int total = NT2 * KS * 512;
    if (e < total) {
        int frag   = e >> 9;          // /512
        int within = e & 511;
        int lane   = within >> 4;
        int h      = within & 15;
        int ntile  = frag / KS;
        int ks     = frag % KS;
        int kk = (h < 8 ? h : h + 8) + ((lane >= 16) ? 8 : 0);
        int k  = ks * 32 + kk;
        int c  = ntile * 16 + (lane & 15);
        float v = (c < C) ? W[(size_t)k * C + c] : 0.0f;
        wpk[e] = f32_to_bf16(v);
    }
    if (e < NT2 * 16) {
        biasPad[e] = (e < C) ? bias[e] : -1.0e30f;
    }
}

// ---------- kernel 1: fused bf16-WMMA GEMM + online log-softmax + target gather ----------
// Per wave: 16 rows. 4 N-tiles per pass: 1 A-fragment feeds 4 WMMAs.
// Softmax state is per-lane over that lane's column stripe (no shuffles in hot loop);
// cross-lane merge done once at the end.
__global__ __launch_bounds__(128) void gemm_ce_kernel(
        const float* __restrict__ feat, const int* __restrict__ targets,
        const unsigned short* __restrict__ wpk, const float* __restrict__ biasPad,
        float* __restrict__ losses, int NT2) {
    constexpr int D  = 512;
    constexpr int KS = D / 32;                       // 16 k-steps
    __shared__ unsigned short ldsA[4 * 16 * D];      // 64 KB: 4 waves x 16 rows x 512 bf16

    const int lane  = threadIdx.x & 31;
    const int wave  = threadIdx.x >> 5;
    const int rowBase = blockIdx.x * 64 + wave * 16;
    const int laneHi  = (lane >= 16) ? 1 : 0;
    const int rofs    = laneHi ? 8 : 0;

    // --- stage A: 16 rows fp32 -> bf16 into this wave's LDS slice (row-major [16][512]) ---
    unsigned short* aw = &ldsA[wave * 16 * D];
    {
        const float4* src = (const float4*)(feat + (size_t)rowBase * D);
        unsigned long long* dst = (unsigned long long*)aw;
        for (int i = lane; i < (16 * D) >> 2; i += 32) {
            float4 f = src[i];
            unsigned long long p =
                  (unsigned long long)f32_to_bf16(f.x)
                | ((unsigned long long)f32_to_bf16(f.y) << 16)
                | ((unsigned long long)f32_to_bf16(f.z) << 32)
                | ((unsigned long long)f32_to_bf16(f.w) << 48);
            dst[i] = p;
        }
    }
    asm volatile("s_wait_dscnt 0" ::: "memory");   // wave-local LDS stores complete before reads

    // --- per-row state; precompute loop-invariant target info ---
    float m[8], s[8], ta[8];
    int   ttile[8];
    bool  mine[8];
#pragma unroll
    for (int r = 0; r < 8; ++r) {
        int tc = targets[rowBase + r + rofs];
        ttile[r] = tc >> 4;
        mine[r]  = (tc & 15) == (lane & 15);
        m[r] = -3.0e38f; s[r] = 0.0f; ta[r] = 0.0f;
    }

    // A base: byte pointer; per k-step offsets are compile-time-affine (immediate offsets)
    const char* abase = (const char*)aw + ((lane & 15) * D + (laneHi ? 8 : 0)) * 2;
    // W base: one advancing pointer; all (j, ks) offsets fold into 24-bit immediates
    const char* wp = (const char*)wpk + lane * 32;
    const float* bp = biasPad + (lane & 15);

    const int nBlk = NT2 >> 2;
    for (int blk = 0; blk < nBlk; ++blk) {
        v8f acc[4];
#pragma unroll
        for (int j = 0; j < 4; ++j) acc[j] = {};

#pragma unroll 4
        for (int ks = 0; ks < KS; ++ks) {
            ABfrag a;
            a.h[0] = *(const us8v*)(abase + ks * 64);
            a.h[1] = *(const us8v*)(abase + ks * 64 + 32);
#pragma unroll
            for (int j = 0; j < 4; ++j) {
                ABfrag b;
                b.h[0] = *(const us8v*)(wp + j * (KS * 1024) + ks * 1024);
                b.h[1] = *(const us8v*)(wp + j * (KS * 1024) + ks * 1024 + 16);
                acc[j] = __builtin_amdgcn_wmma_f32_16x16x32_bf16(
                             false, a.bf, false, b.bf, (short)0, acc[j], false, false);
            }
        }
        wp += 4 * KS * 1024;

#pragma unroll
        for (int j = 0; j < 4; ++j) {
            const int nt = blk * 4 + j;
            const float bias_v = bp[nt * 16];
#pragma unroll
            for (int r = 0; r < 8; ++r) {
                float v = acc[j][r] + bias_v;
                // one-transcendental online logsumexp update (per-lane stripe)
                float d = v - m[r];
                float e = __expf(-fabsf(d));
                s[r] = (d > 0.0f) ? fmaf(s[r], e, 1.0f) : (s[r] + e);
                m[r] = fmaxf(m[r], v);
                if (mine[r] && ttile[r] == nt) ta[r] += v;
            }
        }
    }

    // --- cross-lane merge of (m, s) across the 16-lane half, then write losses ---
#pragma unroll
    for (int r = 0; r < 8; ++r) {
        float mr = m[r], sr = s[r];
        for (int o = 8; o >= 1; o >>= 1) {
            float om = __shfl_xor(mr, o, 16);
            float os = __shfl_xor(sr, o, 16);
            float d  = mr - om;
            float e  = __expf(-fabsf(d));
            sr = (d > 0.0f) ? fmaf(os, e, sr) : fmaf(sr, e, os);
            mr = fmaxf(mr, om);
        }
        float tl   = rsum16(ta[r]);
        float loss = mr + __logf(sr) - tl;
        if ((lane & 15) == r) losses[rowBase + r + rofs] = loss;
    }
}

// ---------- top-k selection kernels ----------
__global__ void zero_kernel(unsigned* __restrict__ p, int n) {
    int i = blockIdx.x * 256 + threadIdx.x;
    if (i < n) p[i] = 0u;
}

__global__ void hist_hi_kernel(const float* __restrict__ losses, unsigned* __restrict__ hist, int n) {
    int i = blockIdx.x * 256 + threadIdx.x;
    if (i < n) atomicAdd(&hist[keyOf(losses[i]) >> 16], 1u);
}

__global__ void hist_lo_kernel(const float* __restrict__ losses, unsigned* __restrict__ hist,
                               const unsigned* __restrict__ ctl, int n) {
    int i = blockIdx.x * 256 + threadIdx.x;
    if (i < n) {
        unsigned key = keyOf(losses[i]);
        if ((key >> 16) == ctl[0]) atomicAdd(&hist[key & 0xFFFFu], 1u);
    }
}

// Finds, scanning bins from high to low, the bin where cumulative count crosses k.
__global__ void select_kernel(const unsigned* __restrict__ hist, unsigned* __restrict__ ctl,
                              int kTotal, int pass) {
    __shared__ unsigned sh[256];
    int t = threadIdx.x;
    unsigned c = 0;
    for (int b = t * 256; b < t * 256 + 256; ++b) c += hist[b];
    sh[t] = c;
    __syncthreads();
    if (t == 0) {
        unsigned k = (pass == 0) ? (unsigned)kTotal : ctl[1];
        unsigned cum = 0; int g = 0;
        for (int gg = 255; gg >= 0; --gg) {
            if (cum + sh[gg] >= k) { g = gg; break; }
            cum += sh[gg];
        }
        unsigned kin = k - cum;
        unsigned cumb = 0; int sel = g * 256;
        for (int b = g * 256 + 255; b >= g * 256; --b) {
            if (cumb + hist[b] >= kin) { sel = b; break; }
            cumb += hist[b];
        }
        unsigned kr = kin - cumb;   // # of threshold-valued elements to include
        if (pass == 0) { ctl[0] = (unsigned)sel; ctl[1] = kr; }
        else           { ctl[2] = (ctl[0] << 16) | (unsigned)sel; ctl[3] = kr; }
    }
}

// Deterministic partial sums of losses strictly above threshold.
__global__ void sum_gt_kernel(const float* __restrict__ losses, const unsigned* __restrict__ ctl,
                              float* __restrict__ partials, int n) {
    __shared__ float sh[256];
    const unsigned T = ctl[2];
    float v = 0.0f;
    for (int i = blockIdx.x * 256 + threadIdx.x; i < n; i += gridDim.x * 256) {
        float x = losses[i];
        if (keyOf(x) > T) v += x;
    }
    sh[threadIdx.x] = v;
    __syncthreads();
    for (int sft = 128; sft > 0; sft >>= 1) {
        if (threadIdx.x < sft) sh[threadIdx.x] += sh[threadIdx.x + sft];
        __syncthreads();
    }
    if (threadIdx.x == 0) partials[blockIdx.x] = sh[0];
}

__global__ void finalize_kernel(const float* __restrict__ partials, const unsigned* __restrict__ ctl,
                                int k, float* __restrict__ out) {
    __shared__ float sh[256];
    sh[threadIdx.x] = partials[threadIdx.x];
    __syncthreads();
    for (int sft = 128; sft > 0; sft >>= 1) {
        if (threadIdx.x < sft) sh[threadIdx.x] += sh[threadIdx.x + sft];
        __syncthreads();
    }
    if (threadIdx.x == 0) {
        float tv = floatOf(ctl[2]);
        out[0] = (sh[0] + (float)ctl[3] * tv) / (float)k;
    }
}

// ---------- launcher ----------
extern "C" void kernel_launch(void* const* d_in, const int* in_sizes, int n_in,
                              void* d_out, int out_size, void* d_ws, size_t ws_size,
                              hipStream_t stream) {
    const float* feat    = (const float*)d_in[0];
    const int*   targets = (const int*)  d_in[1];
    const float* W       = (const float*)d_in[2];
    const float* bias    = (const float*)d_in[3];

    const int B   = in_sizes[1];             // 65536
    const int C   = in_sizes[3];             // 1000
    const int D   = in_sizes[0] / B;         // 512 (kernel assumes 512)
    const int NT  = (C + 15) / 16;           // 63
    const int NT2 = (NT + 3) & ~3;           // 64 (multiple of 4)
    const int KS  = D / 32;                  // 16
    const int k   = (int)((double)B * 0.7);  // 45875

    char* ws = (char*)d_ws;
    size_t off = 0;
    float*          losses  = (float*)(ws + off);           off += (size_t)B * 4;        off = (off + 255) & ~(size_t)255;
    unsigned*       hist    = (unsigned*)(ws + off);        off += 65536u * 4;
    unsigned*       ctl     = (unsigned*)(ws + off);        off += 256;
    float*          partials= (float*)(ws + off);           off += 256 * 4;              off = (off + 255) & ~(size_t)255;
    unsigned short* wpk     = (unsigned short*)(ws + off);  off += (size_t)NT2 * KS * 512 * 2; off = (off + 255) & ~(size_t)255;
    float*          biasPad = (float*)(ws + off);

    // 1) pack W + bias (padded to NT2 tiles)
    {
        int total = NT2 * KS * 512;
        packW_kernel<<<(total + 255) / 256, 256, 0, stream>>>(W, bias, wpk, biasPad, C, NT2, KS);
    }
    // 2) fused GEMM + cross-entropy
    gemm_ce_kernel<<<B / 64, 128, 0, stream>>>(feat, targets, wpk, biasPad, losses, NT2);

    // 3) two-level radix select for k-th largest
    zero_kernel<<<(65536 + 64 + 255) / 256, 256, 0, stream>>>(hist, 65536 + 64); // hist + ctl (contiguous)
    hist_hi_kernel<<<(B + 255) / 256, 256, 0, stream>>>(losses, hist, B);
    select_kernel<<<1, 256, 0, stream>>>(hist, ctl, k, 0);
    zero_kernel<<<65536 / 256, 256, 0, stream>>>(hist, 65536);
    hist_lo_kernel<<<(B + 255) / 256, 256, 0, stream>>>(losses, hist, ctl, B);
    select_kernel<<<1, 256, 0, stream>>>(hist, ctl, k, 1);

    // 4) deterministic mean of top-k
    sum_gt_kernel<<<256, 256, 0, stream>>>(losses, ctl, partials, B);
    finalize_kernel<<<1, 256, 0, stream>>>(partials, ctl, k, (float*)d_out);
}
